// StackedCustomLstm_22582938043099
// MI455X (gfx1250) — matrix operand order repaired and checked by reading, not compiled, not run on hardware
//
#include <hip/hip_runtime.h>

#define B_ 32
#define T_ 512
#define D_ 1024
#define H_ 1024
#define L_ 4
#define G4 4096   // 4*H
#define KC 2048   // D + H rows of combined transposed weight

typedef __attribute__((ext_vector_type(16))) __bf16 v16bf;
typedef __attribute__((ext_vector_type(8)))  float  v8f;
typedef __attribute__((ext_vector_type(4)))  unsigned int u32x4;

union AFrag { unsigned int u[8]; v16bf v; };
union BFrag { u32x4 q[2];        v16bf v; };

__device__ __forceinline__ float sigm(float x) { return 1.0f / (1.0f + __expf(-x)); }

// ---------------------------------------------------------------------------
// CDNA5 async global->LDS staging (ASYNCcnt path, bypasses VGPRs).
// Stages a 32-row x 256-K bf16 tile (16 KB) into an LDS buffer with a
// 528-byte row pitch (132 dwords; pitch%64dw==4 keeps DS banks spread).
// One vector instruction moves 32 lanes x 16 B; 4 issues/thread-block cover
// the tile with a 256-thread block.
// ---------------------------------------------------------------------------
__device__ __forceinline__ void stage_async(const __bf16* __restrict__ src,
                                            int k0, unsigned ldsbase, int tid) {
#pragma unroll
  for (int it = 0; it < 4; ++it) {
    int i = tid + it * 256;          // 0..1023 : 32 rows x 32 chunks of 16B
    int m = i >> 5, c = i & 31;
    const __bf16* gp = src + (size_t)m * 1024 + k0 + c * 8;
    unsigned ld = ldsbase + (unsigned)(m * 528 + c * 16);
    asm volatile("global_load_async_to_lds_b128 %0, %1, off"
                 :: "v"(ld), "v"(gp) : "memory");
  }
}

__device__ __forceinline__ void wait_async0() {
  asm volatile("s_wait_asynccnt 0" ::: "memory");
}

// ---------------------------------------------------------------------------
// One-time conversion kernels
// ---------------------------------------------------------------------------

// Wt[l][k][g] (bf16): k<1024 -> W_ih[l][g][k]; else W_hh[l][g][k-1024]
__global__ void build_wt(const float* __restrict__ Wih, const float* __restrict__ Whh,
                         __bf16* __restrict__ Wt) {
  const size_t total = (size_t)L_ * KC * G4;
  for (size_t i = (size_t)blockIdx.x * blockDim.x + threadIdx.x; i < total;
       i += (size_t)gridDim.x * blockDim.x) {
    int g = (int)(i & (G4 - 1));
    size_t rest = i >> 12;
    int k = (int)(rest & (KC - 1));
    int l = (int)(rest >> 11);
    float v = (k < D_) ? Wih[((size_t)l * G4 + g) * D_ + k]
                       : Whh[((size_t)l * G4 + g) * H_ + (k - D_)];
    Wt[i] = (__bf16)v;
  }
}

__global__ void combine_bias(const float* __restrict__ a, const float* __restrict__ b,
                             float* __restrict__ o) {
  int i = blockIdx.x * blockDim.x + threadIdx.x;
  if (i < L_ * G4) o[i] = a[i] + b[i];
}

// x [B,T,D] fp32 -> seq [T,B,D] bf16 (time-major)
__global__ void x_to_bf16(const float* __restrict__ x, __bf16* __restrict__ s) {
  const size_t total = (size_t)T_ * B_ * D_;
  for (size_t i = (size_t)blockIdx.x * blockDim.x + threadIdx.x; i < total;
       i += (size_t)gridDim.x * blockDim.x) {
    int d = (int)(i & (D_ - 1));
    int b = (int)((i >> 10) & (B_ - 1));
    int t = (int)(i >> 15);
    s[i] = (__bf16)x[((size_t)b * T_ + t) * D_ + d];
  }
}

__global__ void init_state(__bf16* hA, __bf16* hB, float* cS) {
  int i = blockIdx.x * blockDim.x + threadIdx.x;  // 32768 threads
  if (i < B_ * H_) {
    hA[i] = (__bf16)0.0f;
    hB[i] = (__bf16)0.0f;
    cS[i] = 0.0f;
  }
}

// ---------------------------------------------------------------------------
// Batched input projection: xw[t][b][g] = sum_d seq[t][b][d]*Wt[d][g] + bias[g]
// grid = (64 n-chunks, 512 t), block = 256 (8 waves; wave = 16x16 tile).
// A tile double-buffered in LDS via async loads, overlapped with WMMA.
// ---------------------------------------------------------------------------
__global__ __launch_bounds__(256) void proj_gemm(
    const __bf16* __restrict__ seq_in,  // [T,B,D] bf16
    const __bf16* __restrict__ WtL,     // [2048,4096], rows 0..1023 = x part
    const float*  __restrict__ biasL,   // [4096]
    float*        __restrict__ xw)      // [T,B,4096]
{
  const int t = blockIdx.y;
  const int nblk = blockIdx.x;
  const int tid = threadIdx.x;
  const int wave = tid >> 5, lane = tid & 31;
  const int hf = lane >> 4, ln = lane & 15;
  const int mTile = wave & 1, nSub = wave >> 1;
  const int ncol0 = nblk * 64 + nSub * 16;

  __shared__ unsigned int Alds[2][32 * 132];
  const unsigned base0 = (unsigned)(uintptr_t)&Alds[0][0];
  const unsigned base1 = (unsigned)(uintptr_t)&Alds[1][0];

  v8f acc = {0.f, 0.f, 0.f, 0.f, 0.f, 0.f, 0.f, 0.f};
  const __bf16* arow = seq_in + (size_t)t * (B_ * D_);

  stage_async(arow, 0, base0, tid);
  for (int ch = 0; ch < 4; ++ch) {
    wait_async0();       // own async loads of chunk ch complete
    __syncthreads();     // whole cooperative tile visible; prev buffer free
    if (ch + 1 < 4) stage_async(arow, (ch + 1) * 256, (ch & 1) ? base0 : base1, tid);

    const unsigned int* AB = &Alds[ch & 1][0];
    const int k0 = ch * 256;
#pragma unroll
    for (int ks = 0; ks < 8; ++ks) {
      AFrag a;
      const unsigned int* ar = AB + (mTile * 16 + ln) * 132 + ks * 16;
#pragma unroll
      for (int j = 0; j < 8; ++j) {
        // CDNA5 16-bit A layout: vgpr j holds K = (j/4)*16 + half*8 + (j%4)*2 (+1)
        a.u[j] = ar[(j >> 2) * 8 + hf * 4 + (j & 3)];
      }
      BFrag b;  // lane = N column, elements over K in table order
      const __bf16* bp = WtL + (size_t)(k0 + ks * 32 + hf * 8) * G4 + ncol0 + ln;
#pragma unroll
      for (int s = 0; s < 16; ++s) b.v[s] = bp[(size_t)(s < 8 ? s : s + 8) * G4];
      acc = __builtin_amdgcn_wmma_f32_16x16x32_bf16(false, a.v, false, b.v,
                                                    (short)0, acc, false, false);
    }
  }

  const float bv = biasL[ncol0 + ln];
  float* orow = xw + (size_t)t * (B_ * G4);
#pragma unroll
  for (int r = 0; r < 8; ++r) {
    int m = mTile * 16 + hf * 8 + r;  // C/D layout: vgpr r, lanes16-31 -> M+8
    orow[(size_t)m * G4 + ncol0 + ln] = acc[r] + bv;
  }
}

// ---------------------------------------------------------------------------
// One recurrent timestep, fused gates + cell update.
// grid = 64 (each block owns 16 h-columns and its 4 gate slices), block = 256.
// ---------------------------------------------------------------------------
__global__ __launch_bounds__(256) void lstm_step(
    const float*  __restrict__ xw_t,    // [B,4096] precomputed x-part + bias
    const __bf16* __restrict__ WhhT,    // [1024,4096] transposed h-part of Wt
    const __bf16* __restrict__ h_in,    // [B,H] bf16
    __bf16*       __restrict__ h_out,   // [B,H] bf16
    float*        __restrict__ c_st,    // [B,H] fp32 (in/out)
    __bf16*       __restrict__ seq_out, // [B,H] bf16 (next layer input @ t)
    float*        __restrict__ out_t,   // [B,H] fp32 into d_out (final layer)
    float*        __restrict__ hfin,    // [B,H]
    float*        __restrict__ cfin,    // [B,H]
    int is_last, int write_fin)
{
  const int tid = threadIdx.x;
  const int wave = tid >> 5, lane = tid & 31;
  const int hf = lane >> 4, ln = lane & 15;
  const int mTile = wave & 1, gate = wave >> 1;  // gate: 0=i 1=f 2=g 3=o
  const int hblk = blockIdx.x;                   // 0..63
  const int gcol0 = gate * H_ + hblk * 16;

  __shared__ unsigned int Alds[2][32 * 132];
  __shared__ float gld[4][32][16];
  const unsigned base0 = (unsigned)(uintptr_t)&Alds[0][0];
  const unsigned base1 = (unsigned)(uintptr_t)&Alds[1][0];

  v8f acc = {0.f, 0.f, 0.f, 0.f, 0.f, 0.f, 0.f, 0.f};

  stage_async(h_in, 0, base0, tid);
  for (int ch = 0; ch < 4; ++ch) {
    wait_async0();
    __syncthreads();
    if (ch + 1 < 4) stage_async(h_in, (ch + 1) * 256, (ch & 1) ? base0 : base1, tid);

    const unsigned int* AB = &Alds[ch & 1][0];
    const int k0 = ch * 256;
#pragma unroll
    for (int ks = 0; ks < 8; ++ks) {
      AFrag a;
      const unsigned int* ar = AB + (mTile * 16 + ln) * 132 + ks * 16;
#pragma unroll
      for (int j = 0; j < 8; ++j) {
        a.u[j] = ar[(j >> 2) * 8 + hf * 4 + (j & 3)];
      }
      BFrag b;
      const __bf16* bp = WhhT + (size_t)(k0 + ks * 32 + hf * 8) * G4 + gcol0 + ln;
#pragma unroll
      for (int s = 0; s < 16; ++s) b.v[s] = bp[(size_t)(s < 8 ? s : s + 8) * G4];
      acc = __builtin_amdgcn_wmma_f32_16x16x32_bf16(false, a.v, false, b.v,
                                                    (short)0, acc, false, false);
    }
  }

  // gates = recurrent accum + precomputed input projection (+bias)
#pragma unroll
  for (int r = 0; r < 8; ++r) {
    int m = mTile * 16 + hf * 8 + r;
    gld[gate][m][ln] = acc[r] + xw_t[(size_t)m * G4 + gcol0 + ln];
  }
  __syncthreads();

  // fused LSTM cell update for this block's 32x16 slice
  for (int e = tid; e < 32 * 16; e += 256) {
    int m = e >> 4, n = e & 15;
    int col = hblk * 16 + n;
    float gi = gld[0][m][n];
    float gf = gld[1][m][n];
    float gg = gld[2][m][n];
    float go = gld[3][m][n];
    float c_old = c_st[m * H_ + col];
    float c_new = sigm(gf) * c_old + sigm(gi) * tanhf(gg);
    float h = sigm(go) * tanhf(c_new);
    c_st[m * H_ + col] = c_new;
    h_out[m * H_ + col] = (__bf16)h;
    seq_out[m * H_ + col] = (__bf16)h;
    if (is_last) out_t[m * H_ + col] = h;
    if (write_fin) {
      hfin[m * H_ + col] = h;
      cfin[m * H_ + col] = c_new;
    }
  }
}

// pack_padded masking: zero outputs past each sequence length
__global__ void mask_out(float* __restrict__ out, const int* __restrict__ len) {
  size_t i = (size_t)blockIdx.x * blockDim.x + threadIdx.x;
  if (i >= (size_t)T_ * B_ * H_) return;
  int b = (int)((i >> 10) & (B_ - 1));
  int t = (int)(i >> 15);
  if (t >= len[b]) out[i] = 0.0f;
}

// ---------------------------------------------------------------------------

extern "C" void kernel_launch(void* const* d_in, const int* in_sizes, int n_in,
                              void* d_out, int out_size, void* d_ws, size_t ws_size,
                              hipStream_t stream) {
  (void)in_sizes; (void)n_in; (void)out_size; (void)ws_size;

  const float* x       = (const float*)d_in[0];
  const int*   lengths = (const int*)d_in[1];
  const float* Wih     = (const float*)d_in[2];
  const float* Whh     = (const float*)d_in[3];
  const float* bih     = (const float*)d_in[4];
  const float* bhh     = (const float*)d_in[5];
  float* out = (float*)d_out;

  char* ws = (char*)d_ws;
  size_t off = 0;
  auto alloc = [&](size_t bytes) -> void* {
    void* p = ws + off;
    off += (bytes + 255) & ~(size_t)255;
    return p;
  };
  __bf16* Wt    = (__bf16*)alloc((size_t)L_ * KC * G4 * 2);   //  64 MB
  float*  biasC = (float*) alloc((size_t)L_ * G4 * 4);        //  64 KB
  float*  xw    = (float*) alloc((size_t)T_ * B_ * G4 * 4);   // 256 MB
  __bf16* seqA  = (__bf16*)alloc((size_t)T_ * B_ * H_ * 2);   //  32 MB
  __bf16* seqB  = (__bf16*)alloc((size_t)T_ * B_ * H_ * 2);   //  32 MB
  __bf16* hA    = (__bf16*)alloc((size_t)B_ * H_ * 2);
  __bf16* hB    = (__bf16*)alloc((size_t)B_ * H_ * 2);
  float*  cS    = (float*) alloc((size_t)B_ * H_ * 4);

  build_wt<<<4096, 256, 0, stream>>>(Wih, Whh, Wt);
  combine_bias<<<64, 256, 0, stream>>>(bih, bhh, biasC);
  x_to_bf16<<<4096, 256, 0, stream>>>(x, seqA);

  float* hfin_base = out + (size_t)T_ * B_ * H_;
  float* cfin_base = hfin_base + (size_t)L_ * B_ * H_;

  __bf16* in_seq = seqA;
  __bf16* out_seq = seqB;
  for (int l = 0; l < L_; ++l) {
    const __bf16* WtL = Wt + (size_t)l * KC * G4;
    proj_gemm<<<dim3(64, T_), 256, 0, stream>>>(in_seq, WtL, biasC + l * G4, xw);
    init_state<<<128, 256, 0, stream>>>(hA, hB, cS);

    const __bf16* WhhT = WtL + (size_t)D_ * G4;
    __bf16* hin = hA;
    __bf16* hout = hB;
    for (int t = 0; t < T_; ++t) {
      lstm_step<<<64, 256, 0, stream>>>(
          xw + (size_t)t * B_ * G4, WhhT, hin, hout, cS,
          out_seq + (size_t)t * B_ * H_,
          out + (size_t)t * B_ * H_,
          hfin_base + (size_t)l * B_ * H_,
          cfin_base + (size_t)l * B_ * H_,
          (l == L_ - 1) ? 1 : 0, (t == T_ - 1) ? 1 : 0);
      __bf16* tmp = hin; hin = hout; hout = tmp;
    }
    __bf16* ts = in_seq; in_seq = out_seq; out_seq = ts;
  }

  mask_out<<<(T_ * B_ * H_ + 255) / 256, 256, 0, stream>>>(out, lengths);
}
